// MultiHeadAttentionBlock_2190433321704
// MI455X (gfx1250) — compile-verified, matrix operand order checked by reading, not run on hardware
//
#include <hip/hip_runtime.h>
#include <hip/hip_bf16.h>

// ---------------------------------------------------------------------------
// Types
// ---------------------------------------------------------------------------
typedef __attribute__((ext_vector_type(16))) __bf16 v16bf;
typedef __attribute__((ext_vector_type(8)))  __bf16 v8bf;
typedef __attribute__((ext_vector_type(8)))  float  v8f;

#define D_MODEL 1024
#define NHEAD   16
#define DK      64
#define BB      4
#define LL      2048
#define MROWS   (BB * LL)   // 8192

// ---------------------------------------------------------------------------
// Fragment loader. 16-bit A-matrix 16x32 layout (ISA 7.12.2):
//   lane r (0..15) holds row r; lanes 16..31 hold rows 0..15 too (h=1).
//   halves 0..7  come from k = 8h + 0..7
//   halves 8..15 come from k = 8h + 16..23
// B fragments use the symmetric layout (lane n = column n, same K pattern),
// so a row-major [N,K] source loads identically.
// ---------------------------------------------------------------------------
__device__ __forceinline__ v16bf load_frag(const __bf16* base, int ld) {
    const int lane = threadIdx.x & 31;
    const int r = lane & 15, h = lane >> 4;
    const __bf16* row = base + r * ld + 8 * h;
    v8bf lo = *(const v8bf*)(row);
    v8bf hi = *(const v8bf*)(row + 16);
    return __builtin_shufflevector(lo, hi, 0,1,2,3,4,5,6,7,8,9,10,11,12,13,14,15);
}

__device__ __forceinline__ v8f wmma_bf16(v16bf a, v16bf b, v8f c) {
    return __builtin_amdgcn_wmma_f32_16x16x32_bf16(
        /*neg_a=*/false, a, /*neg_b=*/false, b,
        /*c_mod=*/(short)0, c, /*reuse_a=*/false, /*reuse_b=*/false);
}

// ---------------------------------------------------------------------------
// Pre-pass: vectorized f32 -> bf16 (native packed converts)
// ---------------------------------------------------------------------------
__global__ __launch_bounds__(256) void cvt_f32_bf16(const float* __restrict__ src,
                                                    __bf16* __restrict__ dst, int n8) {
    const int idx = blockIdx.x * 256 + threadIdx.x;   // 8 elements per thread
    if (idx < n8) {
        v8f x = *(const v8f*)(src + (size_t)idx * 8);
        v8bf y = __builtin_convertvector(x, v8bf);
        *(v8bf*)(dst + (size_t)idx * 8) = y;
    }
}

// ---------------------------------------------------------------------------
// Stage 1 / 4: out = X @ W^T  (X: [8192,1024] bf16, W: [1024,1024] bf16)
// Each wave: 32(M) x 64(N) tile, K loop in steps of 32 (bf16 WMMA).
// OUT_MODE 0: f32 row-major [8192,1024]
// OUT_MODE 1: bf16 [B,H,L,64]   (head-split, row-major rows = l)
// OUT_MODE 2: bf16 [B,H,64,L]   (head-split, transposed)
// FEAT: apply elu(x)+1 = (x>0 ? x+1 : exp(x)) before store.
// ---------------------------------------------------------------------------
template<bool FEAT, int OUT_MODE>
__global__ __launch_bounds__(256) void gemm_proj(const __bf16* __restrict__ X,
                                                 const __bf16* __restrict__ W,
                                                 void* __restrict__ Out) {
    const int wid = threadIdx.x >> 5;
    const int gw  = blockIdx.x * 8 + wid;   // 4096 waves total
    const int mb  = gw & 255;               // 256 m-blocks of 32 rows
    const int nb  = gw >> 8;                // 16  n-blocks of 64 cols (= head)
    const int m0  = mb * 32;
    const int n0  = nb * 64;

    v8f c[2][4] = {};

    for (int kk = 0; kk < D_MODEL; kk += 32) {
        v16bf a0 = load_frag(X + (size_t)m0 * D_MODEL + kk, D_MODEL);
        v16bf a1 = load_frag(X + (size_t)(m0 + 16) * D_MODEL + kk, D_MODEL);
        v16bf bfr[4];
#pragma unroll
        for (int nt = 0; nt < 4; ++nt)
            bfr[nt] = load_frag(W + (size_t)(n0 + nt * 16) * D_MODEL + kk, D_MODEL);
#pragma unroll
        for (int nt = 0; nt < 4; ++nt) {
            c[0][nt] = wmma_bf16(a0, bfr[nt], c[0][nt]);
            c[1][nt] = wmma_bf16(a1, bfr[nt], c[1][nt]);
        }
    }

    // store
    const int lane = threadIdx.x & 31;
    const int r = lane & 15, h = lane >> 4;
#pragma unroll
    for (int mt = 0; mt < 2; ++mt) {
#pragma unroll
        for (int nt = 0; nt < 4; ++nt) {
#pragma unroll
            for (int i = 0; i < 8; ++i) {
                const int row = m0 + mt * 16 + i + 8 * h;   // (b, l)
                const int col = n0 + nt * 16 + r;           // feature index
                float v = c[mt][nt][i];
                if (FEAT) v = (v > 0.0f) ? (v + 1.0f) : __expf(v);
                if (OUT_MODE == 0) {
                    ((float*)Out)[(size_t)row * D_MODEL + col] = v;
                } else {
                    const int b = row >> 11;        // /2048
                    const int l = row & (LL - 1);
                    const int d = col & (DK - 1);
                    if (OUT_MODE == 1) {
                        ((__bf16*)Out)[(((size_t)(b * NHEAD + nb)) * LL + l) * DK + d] = (__bf16)v;
                    } else {
                        ((__bf16*)Out)[(((size_t)(b * NHEAD + nb)) * DK + d) * LL + l] = (__bf16)v;
                    }
                }
            }
        }
    }
}

// ---------------------------------------------------------------------------
// Stage 2: KV = K^T V per (b,h).  Kt,Vt: [64, 2048] bf16 per head.
// Output KVc = (K^T V)^T bf16 [64,64] per head (row n holds KV[:,n]),
// so stage-3 B fragments load contiguously.
// One block per (b,h); 8 waves; each wave: one 16x32 output slab (2 tiles).
// ---------------------------------------------------------------------------
__global__ __launch_bounds__(256) void kv_kernel(const __bf16* __restrict__ Kt,
                                                 const __bf16* __restrict__ Vt,
                                                 __bf16* __restrict__ KVc) {
    const int bh  = blockIdx.x;
    const int wid = threadIdx.x >> 5;
    const int mt  = wid & 3;     // M tile (d_k of K)
    const int ntb = wid >> 2;    // 0..1 -> N tiles {2*ntb, 2*ntb+1}

    const __bf16* Kp = Kt + (size_t)bh * DK * LL;
    const __bf16* Vp = Vt + (size_t)bh * DK * LL;

    v8f c0 = {}, c1 = {};
    for (int kk = 0; kk < LL; kk += 32) {
        v16bf a  = load_frag(Kp + (size_t)(mt * 16) * LL + kk, LL);
        v16bf b0 = load_frag(Vp + (size_t)(ntb * 32) * LL + kk, LL);
        v16bf b1 = load_frag(Vp + (size_t)(ntb * 32 + 16) * LL + kk, LL);
        c0 = wmma_bf16(a, b0, c0);
        c1 = wmma_bf16(a, b1, c1);
    }

    const int lane = threadIdx.x & 31;
    const int r = lane & 15, h = lane >> 4;
    __bf16* outp = KVc + (size_t)bh * DK * DK;
#pragma unroll
    for (int i = 0; i < 8; ++i) {
        const int d  = mt * 16 + i + 8 * h;          // M (d_k index)
        const int n0 = ntb * 32 + r;                 // N (d_v index), tile 0
        const int n1 = ntb * 32 + 16 + r;            // tile 1
        outp[(size_t)n0 * DK + d] = (__bf16)c0[i];   // store transposed
        outp[(size_t)n1 * DK + d] = (__bf16)c1[i];
    }
}

// ---------------------------------------------------------------------------
// Stage 2b: Ksum[bh][d] = sum_l Kt[bh][d][l]   (vectorized bf16 reads)
// ---------------------------------------------------------------------------
__global__ __launch_bounds__(256) void ksum_kernel(const __bf16* __restrict__ Kt,
                                                   float* __restrict__ Ksum) {
    const int bh = blockIdx.x;
    const int t  = threadIdx.x;
    const int d  = t & 63;
    const int q  = t >> 6;                           // 0..3
    const __bf16* row = Kt + ((size_t)bh * DK + d) * LL + q * 512;
    float s = 0.0f;
    for (int i = 0; i < 512; i += 8) {
        v8bf x = *(const v8bf*)(row + i);
        v8f  f = __builtin_convertvector(x, v8f);
#pragma unroll
        for (int j = 0; j < 8; ++j) s += f[j];
    }
    __shared__ float red[256];
    red[t] = s;
    __syncthreads();
    if (q == 0)
        Ksum[(size_t)bh * DK + d] = red[d] + red[d + 64] + red[d + 128] + red[d + 192];
}

// ---------------------------------------------------------------------------
// Stage 3: out = (Q @ KV) * Z per (b,h); Z = 1/(Q . Ksum + eps).
// Each wave: 32 rows x 64 cols; K = 64 (2 steps).  Writes bf16 concat
// layout Oc[B, L, H*64] for the final projection.
// ---------------------------------------------------------------------------
__global__ __launch_bounds__(256) void out_kernel(const __bf16* __restrict__ Qf,
                                                  const __bf16* __restrict__ KVc,
                                                  const float* __restrict__ Ksum,
                                                  __bf16* __restrict__ Oc) {
    const int wid = threadIdx.x >> 5;
    const int gw  = blockIdx.x * 8 + wid;   // 4096 waves
    const int bh  = gw >> 6;                // 64 heads total
    const int rb  = gw & 63;                // 64 row blocks of 32
    const int m0  = rb * 32;

    const __bf16* Qp  = Qf  + (size_t)bh * LL * DK;
    const __bf16* KVp = KVc + (size_t)bh * DK * DK;

    v8f c[2][4] = {};
#pragma unroll
    for (int ksx = 0; ksx < 2; ++ksx) {
        const int kk = ksx * 32;
        v16bf a0 = load_frag(Qp + (size_t)m0 * DK + kk, DK);
        v16bf a1 = load_frag(Qp + (size_t)(m0 + 16) * DK + kk, DK);
#pragma unroll
        for (int nt = 0; nt < 4; ++nt) {
            v16bf b = load_frag(KVp + (size_t)(nt * 16) * DK + kk, DK);
            c[0][nt] = wmma_bf16(a0, b, c[0][nt]);
            c[1][nt] = wmma_bf16(a1, b, c[1][nt]);
        }
    }

    // per-row normalizer: lane t owns row m0+t (t = 0..31)
    const int lane = threadIdx.x & 31;
    const float*  ks = Ksum + (size_t)bh * DK;
    const __bf16* qr = Qp + (size_t)(m0 + lane) * DK;
    float s = 0.0f;
#pragma unroll
    for (int d8 = 0; d8 < 8; ++d8) {
        v8bf qv = *(const v8bf*)(qr + d8 * 8);
        v8f  qf = __builtin_convertvector(qv, v8f);
        v8f  kv = *(const v8f*)(ks + d8 * 8);
#pragma unroll
        for (int j = 0; j < 8; ++j) s += qf[j] * kv[j];
    }
    const float z = 1.0f / (s + 1e-8f);

    const int r = lane & 15, h = lane >> 4;
    const int b  = bh >> 4;
    const int hh = bh & 15;
#pragma unroll
    for (int mt = 0; mt < 2; ++mt) {
#pragma unroll
        for (int i = 0; i < 8; ++i) {
            const int rloc = mt * 16 + i + 8 * h;           // row within strip
            const float zz = __shfl(z, rloc, 32);
            const int l = m0 + rloc;
#pragma unroll
            for (int nt = 0; nt < 4; ++nt) {
                const int col = hh * DK + nt * 16 + r;
                Oc[((size_t)(b * LL + l)) * D_MODEL + col] = (__bf16)(c[mt][nt][i] * zz);
            }
        }
    }
}

// ---------------------------------------------------------------------------
// Launch
// ---------------------------------------------------------------------------
extern "C" void kernel_launch(void* const* d_in, const int* in_sizes, int n_in,
                              void* d_out, int out_size, void* d_ws, size_t ws_size,
                              hipStream_t stream) {
    const float* q   = (const float*)d_in[0];
    const float* k   = (const float*)d_in[1];
    const float* v   = (const float*)d_in[2];
    // d_in[3] = mask: all-ones, unused by the linear-attention branch
    const float* w_q = (const float*)d_in[4];
    const float* w_k = (const float*)d_in[5];
    const float* w_v = (const float*)d_in[6];
    const float* w_o = (const float*)d_in[7];
    float* out = (float*)d_out;

    char* ws = (char*)d_ws;
    const size_t HEADBUF = (size_t)BB * NHEAD * LL * DK * sizeof(__bf16); // 16 MB
    __bf16* Qf  = (__bf16*)(ws);                        // [B,H,L,64]
    __bf16* Kt  = (__bf16*)(ws + HEADBUF);              // [B,H,64,L]
    __bf16* Vt  = (__bf16*)(ws + 2 * HEADBUF);          // [B,H,64,L]
    __bf16* Oc  = (__bf16*)(ws + 3 * HEADBUF);          // [B,L,1024]; doubles as Xbf staging
    __bf16* Xbf = Oc;                                   // alias: dead before Oc is written
    __bf16* KVc = (__bf16*)(ws + 4 * HEADBUF);          // [BH,64,64]
    float*  Ksm = (float*)(ws + 4 * HEADBUF + (size_t)BB * NHEAD * DK * DK * sizeof(__bf16));
    __bf16* Wbf = (__bf16*)((char*)Ksm + (size_t)BB * NHEAD * DK * sizeof(float)); // [1024,1024]

    const int NX8 = MROWS * D_MODEL / 8;     // 1,048,576 -> 4096 blocks
    const int NW8 = D_MODEL * D_MODEL / 8;   //   131,072 ->  512 blocks
    dim3 blk(256);

    // Q projection
    cvt_f32_bf16<<<NX8 / 256, blk, 0, stream>>>(q,   Xbf, NX8);
    cvt_f32_bf16<<<NW8 / 256, blk, 0, stream>>>(w_q, Wbf, NW8);
    gemm_proj<true, 1><<<512, blk, 0, stream>>>(Xbf, Wbf, Qf);
    // K projection
    cvt_f32_bf16<<<NX8 / 256, blk, 0, stream>>>(k,   Xbf, NX8);
    cvt_f32_bf16<<<NW8 / 256, blk, 0, stream>>>(w_k, Wbf, NW8);
    gemm_proj<true, 2><<<512, blk, 0, stream>>>(Xbf, Wbf, Kt);
    // V projection
    cvt_f32_bf16<<<NX8 / 256, blk, 0, stream>>>(v,   Xbf, NX8);
    cvt_f32_bf16<<<NW8 / 256, blk, 0, stream>>>(w_v, Wbf, NW8);
    gemm_proj<false, 2><<<512, blk, 0, stream>>>(Xbf, Wbf, Vt);
    // Linear-attention core
    kv_kernel  <<<BB * NHEAD, blk, 0, stream>>>(Kt, Vt, KVc);
    ksum_kernel<<<BB * NHEAD, blk, 0, stream>>>(Kt, Ksm);
    out_kernel <<<512, blk, 0, stream>>>(Qf, KVc, Ksm, Oc);
    // Output projection (bf16 in, f32 out)
    cvt_f32_bf16<<<NW8 / 256, blk, 0, stream>>>(w_o, Wbf, NW8);
    gemm_proj<false, 0><<<512, blk, 0, stream>>>(Oc, Wbf, out);
}
